// MultidimensionalLSTM_867583394659
// MI455X (gfx1250) — compile-verified, matrix operand order checked by reading, not run on hardware
//
#include <hip/hip_runtime.h>

// ---------------------------------------------------------------------------
// LSTM (relu candidate / relu cell activation) + final dense, for MI455X.
//   B=2048, T=60, F=89, H=1024 (4H=4096), OUT=30
// Strategy:
//   * recurrence is independent per batch row -> 32 blocks x 64 rows, whole
//     T-loop in one kernel, c-state in VGPRs, h-state in LDS (bf16)
//   * combined weight [U; W; pad] (K=1120) in bf16, pre-swizzled into the
//     per-lane WMMA B-fragment layout so each lane does contiguous 128b loads
//   * each wave accumulates i/f/g/o gate tiles for the same h-columns in
//     registers sharing one A fragment -> gate math fully in registers
//   * an asm-opaque ZERO OFFSET (not pointer!) per timestep blocks LICM from
//     hoisting the t-invariant weight loads (which spilled to scratch) while
//     keeping address-space inference -> global_load_b128, not flat_load
//   * one-stage double buffering of B fragments hides L2 latency
// ---------------------------------------------------------------------------

typedef __attribute__((ext_vector_type(16))) __bf16        v16bf;
typedef __attribute__((ext_vector_type(8)))  float         v8f;
typedef __attribute__((ext_vector_type(4)))  unsigned int  u32x4;

#define B_SZ   2048
#define T_SZ   60
#define F_SZ   89
#define H_SZ   1024
#define G_SZ   4096      // 4*H
#define OUT_SZ 30
#define KTOT   1120      // H + F padded to multiple of 32
#define NKBLK  35        // KTOT / 32
#define TILEB  64        // batch rows per block
#define NBLK   (B_SZ / TILEB)   // 32 blocks
#define LDSTR  1128      // bf16 row stride in LDS (1120 + 8 pad -> bank-conflict free)
#define NTIL_N 256       // 4096 / 16 column tiles
#define GOFF(g) ((size_t)(g) * 64 * NKBLK * 1024)   // gate stride in swizzled weights

union FragB16 { v16bf v; u32x4 u[2]; };

static __device__ inline float sigmoidf_fast(float z) {
  return 1.0f / (1.0f + __expf(-z));
}

// ---------------------------------------------------------------------------
// Pre-pass: build combined [U; W; 0] (K=1120 x N=4096) in bf16, laid out in
// WMMA B-fragment order: [ntile][kblk][lane][16 bf16]  (32 bytes per lane).
// Element e of lane l maps to K = kblk*32 + (e>=8 ? 16:0) + (l>=16 ? 8:0) + (e&7),
// N = ntile*16 + (l&15)  -- mirror of the documented 16-bit A-fragment layout.
// ---------------------------------------------------------------------------
__global__ __launch_bounds__(256) void uswz_kernel(const float* __restrict__ W,
                                                   const float* __restrict__ U,
                                                   unsigned short* __restrict__ uswz) {
  int idx = blockIdx.x * 256 + threadIdx.x;
  if (idx >= NTIL_N * NKBLK * 32) return;
  int lane  = idx & 31;
  int t2    = idx >> 5;
  int kblk  = t2 % NKBLK;
  int ntile = t2 / NKBLK;
  int n     = ntile * 16 + (lane & 15);
  int khalf = (lane >> 4) << 3;

  unsigned int q[8];
#pragma unroll
  for (int j = 0; j < 8; ++j) {
    unsigned int packed = 0;
#pragma unroll
    for (int h = 0; h < 2; ++h) {
      int e = j * 2 + h;
      int k = kblk * 32 + ((e >> 3) << 4) + khalf + (e & 7);
      float v = 0.0f;
      if (k < H_SZ)                 v = U[k * G_SZ + n];
      else if ((k - H_SZ) < F_SZ)   v = W[(k - H_SZ) * G_SZ + n];
      unsigned int bits = __float_as_uint(v);
      unsigned int bf   = (bits + 0x7FFFu + ((bits >> 16) & 1u)) >> 16;  // RNE
      packed |= (bf & 0xFFFFu) << (16 * h);
    }
    q[j] = packed;
  }
  u32x4 o0 = {q[0], q[1], q[2], q[3]};
  u32x4 o1 = {q[4], q[5], q[6], q[7]};
  u32x4* dst = (u32x4*)(uswz + (size_t)idx * 16);
  dst[0] = o0;
  dst[1] = o1;
}

// ---------------------------------------------------------------------------
// Main persistent-recurrence kernel. 512 threads = 16 waves:
//   wave = 4*nw + mw ; mw in 0..3 owns M rows [mw*16, mw*16+16)
//   nw in 0..3 owns h-columns [nw*256, nw*256+256) = 16 tiles of 16
// ---------------------------------------------------------------------------
__global__ __launch_bounds__(512, 1) void lstm_kernel(
    const float* __restrict__ x,     // [B, T, F]
    const float* __restrict__ bvec,  // [4H]
    const unsigned short* __restrict__ uswz,
    const float* __restrict__ Wd,    // [H, OUT]
    const float* __restrict__ bd,    // [OUT]
    float* __restrict__ out)         // [B, OUT]
{
  __shared__ __bf16 hx[TILEB * LDSTR];   // 64 x (1024 h | 96 x_t | 8 pad), bf16

  const int tid   = threadIdx.x;
  const int lane  = tid & 31;
  const int wave  = tid >> 5;
  const int mw    = wave & 3;
  const int nw    = wave >> 2;
  const int lm    = lane & 15;
  const int lh    = lane >> 4;           // 0/1 half-wave
  const int bbase = blockIdx.x * TILEB;

  // h0 = 0
  for (int i = tid; i < TILEB * H_SZ; i += 512)
    hx[(i >> 10) * LDSTR + (i & (H_SZ - 1))] = (__bf16)0.0f;

  // cell state in registers: 16 h-tiles x 8 rows
  v8f cst[16];
  const v8f vzero = {0.f, 0.f, 0.f, 0.f, 0.f, 0.f, 0.f, 0.f};
#pragma unroll
  for (int h = 0; h < 16; ++h) cst[h] = vzero;

  // A-fragment base: row (mw*16+lm), K chunk starts at lh*8 elements
  const char* abase   = (const char*)hx + (size_t)((mw * 16 + lm) * LDSTR + lh * 8) * 2;
  // B-fragment base: 32 contiguous bytes per lane; 1024 B per (ntile,kblk)
  const char* bfrbase = (const char*)uswz + (size_t)lane * 32;

  for (int t = 0; t < T_SZ; ++t) {
    // stage bf16(x_t) into columns [1024, 1120) (cols >= F zero-padded)
    for (int i = tid; i < TILEB * 96; i += 512) {
      int row = i / 96, col = i - row * 96;
      float v = (col < F_SZ) ? x[(bbase + row) * (T_SZ * F_SZ) + t * F_SZ + col] : 0.0f;
      hx[row * LDSTR + H_SZ + col] = (__bf16)v;
    }
    __syncthreads();

    // Opaque ZERO offset, redefined every timestep: the weight/bias loads
    // below are t-invariant; without this, LLVM hoists ~9 MB of fragments
    // out of the t-loop and spills to scratch. Using an integer (not the
    // pointer) keeps address-space inference -> global_load, not flat_load.
    size_t zoff = 0;
    asm volatile("" : "+v"(zoff));
    const char*  bfr_t  = bfrbase + zoff;
    const float* bvec_t = (const float*)((const char*)bvec + zoff);

#pragma unroll
    for (int ht = 0; ht < 16; ++ht) {
      const int hcol = nw * 256 + ht * 16 + lm;     // this lane's h column
      // zero-init accumulators (SRC2 inline 0); bias folded into epilogue
      v8f acc0 = vzero, acc1 = vzero, acc2 = vzero, acc3 = vzero;

      const int   ntb = nw * 16 + ht;
      const char* pa  = abase;
      const char* pb  = bfr_t + (size_t)(ntb * NKBLK) * 1024;

      FragB16 a, b0, b1, b2, b3, n0, n1, n2, n3;
      // prologue: prefetch kb = 0
      n0.u[0] = *(const u32x4*)(pb + GOFF(0));  n0.u[1] = *(const u32x4*)(pb + GOFF(0) + 16);
      n1.u[0] = *(const u32x4*)(pb + GOFF(1));  n1.u[1] = *(const u32x4*)(pb + GOFF(1) + 16);
      n2.u[0] = *(const u32x4*)(pb + GOFF(2));  n2.u[1] = *(const u32x4*)(pb + GOFF(2) + 16);
      n3.u[0] = *(const u32x4*)(pb + GOFF(3));  n3.u[1] = *(const u32x4*)(pb + GOFF(3) + 16);

#pragma unroll 2
      for (int kb = 0; kb < NKBLK - 1; ++kb) {
        b0 = n0; b1 = n1; b2 = n2; b3 = n3;
        const char* pn = pb + (size_t)(kb + 1) * 1024;     // prefetch kb+1
        n0.u[0] = *(const u32x4*)(pn + GOFF(0));  n0.u[1] = *(const u32x4*)(pn + GOFF(0) + 16);
        n1.u[0] = *(const u32x4*)(pn + GOFF(1));  n1.u[1] = *(const u32x4*)(pn + GOFF(1) + 16);
        n2.u[0] = *(const u32x4*)(pn + GOFF(2));  n2.u[1] = *(const u32x4*)(pn + GOFF(2) + 16);
        n3.u[0] = *(const u32x4*)(pn + GOFF(3));  n3.u[1] = *(const u32x4*)(pn + GOFF(3) + 16);

        a.u[0] = *(const u32x4*)(pa);
        a.u[1] = *(const u32x4*)(pa + 32);
        pa += 64;

        acc0 = __builtin_amdgcn_wmma_f32_16x16x32_bf16(false, a.v, false, b0.v, (short)0, acc0, false, false);
        acc1 = __builtin_amdgcn_wmma_f32_16x16x32_bf16(false, a.v, false, b1.v, (short)0, acc1, false, false);
        acc2 = __builtin_amdgcn_wmma_f32_16x16x32_bf16(false, a.v, false, b2.v, (short)0, acc2, false, false);
        acc3 = __builtin_amdgcn_wmma_f32_16x16x32_bf16(false, a.v, false, b3.v, (short)0, acc3, false, false);
      }
      // epilogue iteration kb = NKBLK-1 (no prefetch)
      a.u[0] = *(const u32x4*)(pa);
      a.u[1] = *(const u32x4*)(pa + 32);
      acc0 = __builtin_amdgcn_wmma_f32_16x16x32_bf16(false, a.v, false, n0.v, (short)0, acc0, false, false);
      acc1 = __builtin_amdgcn_wmma_f32_16x16x32_bf16(false, a.v, false, n1.v, (short)0, acc1, false, false);
      acc2 = __builtin_amdgcn_wmma_f32_16x16x32_bf16(false, a.v, false, n2.v, (short)0, acc2, false, false);
      acc3 = __builtin_amdgcn_wmma_f32_16x16x32_bf16(false, a.v, false, n3.v, (short)0, acc3, false, false);

      // gate math fully in registers (bias folded here); c stays in VGPRs
      const float bi0 = bvec_t[0 * H_SZ + hcol];
      const float bi1 = bvec_t[1 * H_SZ + hcol];
      const float bi2 = bvec_t[2 * H_SZ + hcol];
      const float bi3 = bvec_t[3 * H_SZ + hcol];
#pragma unroll
      for (int r = 0; r < 8; ++r) {
        float iv = sigmoidf_fast(acc0[r] + bi0);
        float fv = sigmoidf_fast(acc1[r] + bi1);
        float gv = fmaxf(acc2[r] + bi2, 0.0f);   // activation='relu' on candidate
        float ov = sigmoidf_fast(acc3[r] + bi3);
        float cc = fv * cst[ht][r] + iv * gv;
        cst[ht][r] = cc;
        float hv = ov * fmaxf(cc, 0.0f);         // activation='relu' on cell output
        int rowm = mw * 16 + r + lh * 8;         // C/D layout: VGPR r, lanes>=16 -> M+8
        hx[rowm * LDSTR + hcol] = (__bf16)hv;
      }
    }
    __syncthreads();
  }

  // final dense: y = h_T @ Wd + bd  (tiny: 64x30 per block)
  for (int idx = tid; idx < TILEB * OUT_SZ; idx += 512) {
    int m = idx / OUT_SZ, o = idx - m * OUT_SZ;
    float s = bd[o];
    for (int k = 0; k < H_SZ; ++k)
      s = fmaf((float)hx[m * LDSTR + k], Wd[k * OUT_SZ + o], s);
    out[(bbase + m) * OUT_SZ + o] = s;
  }
}

// ---------------------------------------------------------------------------
extern "C" void kernel_launch(void* const* d_in, const int* in_sizes, int n_in,
                              void* d_out, int out_size, void* d_ws, size_t ws_size,
                              hipStream_t stream) {
  (void)in_sizes; (void)n_in; (void)out_size; (void)ws_size;
  const float* x  = (const float*)d_in[0];   // [B,T,F]
  const float* W  = (const float*)d_in[1];   // [F,4H]
  const float* U  = (const float*)d_in[2];   // [H,4H]
  const float* b  = (const float*)d_in[3];   // [4H]
  const float* Wd = (const float*)d_in[4];   // [H,OUT]
  const float* bd = (const float*)d_in[5];   // [OUT]
  float* out = (float*)d_out;

  unsigned short* uswz = (unsigned short*)d_ws;  // 256*35*32*16 bf16 = ~9.2 MB

  int total = NTIL_N * NKBLK * 32;
  uswz_kernel<<<(total + 255) / 256, 256, 0, stream>>>(W, U, uswz);
  lstm_kernel<<<NBLK, 512, 0, stream>>>(x, b, uswz, Wd, bd, out);
}
